// ROI_final_Classifier_11364483465302
// MI455X (gfx1250) — matrix operand order repaired and checked by reading, not compiled, not run on hardware
//
#include <hip/hip_runtime.h>
#include <hip/hip_bf16.h>

typedef __attribute__((ext_vector_type(16))) _Float16 v16h;
typedef __attribute__((ext_vector_type(8)))  float    v8f;

// ---------------------------------------------------------------------------
// Conv (implicit GEMM) + bias + BN + ReLU.
// K-dimension is the *padded* footprint KK' (32 for 5x5, 16 for 3x3) so each
// 32-wide K chunk covers exactly 1 (or 2) input channels and the per-lane
// im2col decode is fully hoisted out of the K loop.
// Block = 256 threads = 8 waves; the 16x32 f16 A chunk is staged in LDS and
// shared by all 8 waves; each wave owns one 16-pixel N tile.
// A/B fragments use contiguous k order (k = 16*lhi + element) for V_WMMA_F32_16X16X32_F16.
// ---------------------------------------------------------------------------
template <int KS, int STRIDE>
__global__ __launch_bounds__(256) void conv_bn_relu_wmma(
    const float* __restrict__ in, const float* __restrict__ w,
    const float* __restrict__ bias, const float* __restrict__ gamma,
    const float* __restrict__ bnb, const float* __restrict__ mu,
    const float* __restrict__ var, float* __restrict__ out,
    int B, int Cin, int Hin, int Win, int Cout, int Hout, int Wout)
{
    constexpr int KK   = KS * KS;              // real footprint (25 or 9)
    constexpr int KKp  = (KS == 5) ? 32 : 16;  // padded footprint
    constexpr int CIPC = 32 / KKp;             // input channels per K chunk
    constexpr int PAD  = (KS - 1) / 2;

    __shared__ _Float16 As[16][34];            // 16 couts x 32 padded-K

    const int tid  = threadIdx.x;
    const int wave = tid >> 5;
    const int lane = tid & 31;
    const int llo  = lane & 15;
    const int lhi  = lane >> 4;

    const int HW     = Hout * Wout;
    const int Npix   = B * HW;
    const int HinWin = Hin * Win;

    const int  n      = (blockIdx.x * 8 + wave) * 16 + llo;
    const bool nvalid = (n < Npix);
    const int  nc     = nvalid ? n : 0;
    const int  b      = nc / HW;
    const int  rr     = nc - b * HW;
    const int  oy     = rr / Wout;
    const int  ox     = rr - oy * Wout;
    const int  iy0    = oy * STRIDE - PAD;
    const int  ix0    = ox * STRIDE - PAD;
    const float* __restrict__ inb = in + (size_t)b * Cin * HinWin;

    const int mbase = blockIdx.y * 16;

    // ---- per-lane B-side im2col decode, hoisted out of the K loop ----
    int      poff[16];
    unsigned okm = 0u;
    #pragma unroll
    for (int e = 0; e < 16; ++e) {
        const int kl  = (lhi << 4) + e;   // padded-K position within chunk
        const int cis = kl / KKp;         // channel sub-index (0 or 1)
        const int kk  = kl - cis * KKp;   // padded footprint position
        const int ky  = kk / KS;
        const int kx  = kk - ky * KS;
        const int iy  = iy0 + ky;
        const int ix  = ix0 + kx;
        const bool ok = nvalid & (kk < KK) &
                        ((unsigned)iy < (unsigned)Hin) &
                        ((unsigned)ix < (unsigned)Win);
        poff[e] = ok ? (cis * HinWin + iy * Win + ix) : 0;
        okm |= (unsigned)ok << e;
    }

    // ---- per-thread A-staging constants: thread -> (row am, padded pos ap) ----
    const int am  = tid >> 4;
    const int ap  = (tid & 15) << 1;          // even padded position
    const int aci = ap / KKp;
    const int akk = ap - aci * KKp;
    const bool aok0 = (akk < KK);
    const bool aok1 = (akk + 1 < KK);
    const int aoff0 = aok0 ? (aci * KK + akk)     : 0;   // clamped offsets
    const int aoff1 = aok1 ? (aci * KK + akk + 1) : 0;
    const float* __restrict__ wrowA = w + (size_t)(mbase + am) * (Cin * KK);

    v8f acc = {};

    for (int ci0 = 0; ci0 < Cin; ci0 += CIPC) {
        // ---- stage 16 x 32 padded A chunk into LDS (f32 -> f16) ----
        {
            const size_t wo = (size_t)ci0 * KK;
            float w0 = wrowA[wo + aoff0];     // unconditional loads
            float w1 = wrowA[wo + aoff1];
            if (!aok0) w0 = 0.0f;
            if (!aok1) w1 = 0.0f;
            As[am][ap]     = (_Float16)w0;
            As[am][ap + 1] = (_Float16)w1;
        }
        __syncthreads();

        // ---- A fragment from LDS: row M = llo, K = 16*lhi + element (contiguous) ----
        v16h afr;
        #pragma unroll
        for (int v = 0; v < 8; ++v) {
            const int k0 = (lhi << 4) + (v << 1);
            afr[2 * v]     = As[llo][k0];
            afr[2 * v + 1] = As[llo][k0 + 1];
        }

        // ---- B fragment: 16 base+offset gathers, mask applied post-load ----
        const float* __restrict__ basep = inb + (size_t)ci0 * HinWin;
        __builtin_prefetch(basep + (size_t)CIPC * HinWin, 0, 1);  // next chunk
        float raw[16];
        #pragma unroll
        for (int e = 0; e < 16; ++e) raw[e] = basep[poff[e]];
        v16h bfr;
        #pragma unroll
        for (int e = 0; e < 16; ++e) {
            const float vv = ((okm >> e) & 1u) ? raw[e] : 0.0f;
            bfr[e] = (_Float16)vv;
        }

        // D = A*B + C, f32 accumulate
        acc = __builtin_amdgcn_wmma_f32_16x16x32_f16(
            /*neg_a=*/false, afr, /*neg_b=*/false, bfr,
            /*c_mod=*/(short)0, acc, /*reuse_a=*/false, /*reuse_b=*/false);

        __syncthreads();   // As reuse next iteration
    }

    if (nvalid) {
        #pragma unroll
        for (int v = 0; v < 8; ++v) {
            const int co  = mbase + (lhi << 3) + v;
            const float inv = gamma[co] * rsqrtf(var[co] + 1e-5f);
            float val = (acc[v] + bias[co]) * inv + (bnb[co] - mu[co] * inv);
            val = fmaxf(val, 0.0f);
            out[((size_t)(b * Cout + co) * Hout + oy) * Wout + ox] = val;
        }
    }
}

// ---------------------------------------------------------------------------
// 2x2 maxpool, pad=1 both sides, stride ps:  Hout = Hin/ps + 1
// ---------------------------------------------------------------------------
__global__ __launch_bounds__(256) void maxpool2x2(
    const float* __restrict__ in, float* __restrict__ out,
    int BC, int Hin, int Win, int Hout, int Wout, int ps)
{
    const int idx = blockIdx.x * blockDim.x + threadIdx.x;
    const int total = BC * Hout * Wout;
    if (idx >= total) return;
    const int x  = idx % Wout;
    int t        = idx / Wout;
    const int y  = t % Hout;
    const int bc = t / Hout;
    const float* __restrict__ p = in + (size_t)bc * Hin * Win;
    const int ys = y * ps - 1;
    const int xs = x * ps - 1;
    float m = -__builtin_inff();
    #pragma unroll
    for (int dy = 0; dy < 2; ++dy) {
        const int iy = ys + dy;
        if ((unsigned)iy >= (unsigned)Hin) continue;
        #pragma unroll
        for (int dx = 0; dx < 2; ++dx) {
            const int ix = xs + dx;
            if ((unsigned)ix >= (unsigned)Win) continue;
            m = fmaxf(m, p[iy * Win + ix]);
        }
    }
    out[idx] = m;
}

// ---------------------------------------------------------------------------
// FC:  y[r,o] = act( x[r,:] . w[o,:] + b[o] ),  one wave32 per output element,
// float4 main loop -> global_load_b128.
// ---------------------------------------------------------------------------
__global__ __launch_bounds__(256) void fc_warp(
    const float* __restrict__ x, const float* __restrict__ w,
    const float* __restrict__ bias, float* __restrict__ y,
    int rows, int din, int dout, int relu)
{
    const int warp = blockIdx.x * (blockDim.x >> 5) + (threadIdx.x >> 5);
    const int lane = threadIdx.x & 31;
    const int total = rows * dout;
    if (warp >= total) return;
    const int r = warp / dout;
    const int o = warp - r * dout;
    const float* __restrict__ xr = x + (size_t)r * din;
    const float* __restrict__ wr = w + (size_t)o * din;

    float s = 0.0f;
    const int din4 = din >> 2;
    const float4* __restrict__ x4 = (const float4*)xr;
    const float4* __restrict__ w4 = (const float4*)wr;
    for (int k = lane; k < din4; k += 32) {
        const float4 a = x4[k];
        const float4 q = w4[k];
        s = fmaf(a.x, q.x, s);
        s = fmaf(a.y, q.y, s);
        s = fmaf(a.z, q.z, s);
        s = fmaf(a.w, q.w, s);
    }
    for (int k = (din4 << 2) + lane; k < din; k += 32) s = fmaf(xr[k], wr[k], s);

    #pragma unroll
    for (int off = 16; off > 0; off >>= 1) s += __shfl_down(s, off, 32);
    if (lane == 0) {
        s += bias[o];
        if (relu) s = fmaxf(s, 0.0f);
        y[(size_t)r * dout + o] = s;
    }
}

// ---------------------------------------------------------------------------
// ROIs: r = int(clip(bb @ M(s,h), 0, s)) for the 3 scales (96 values total)
// ---------------------------------------------------------------------------
__global__ void rois_kernel(const float* __restrict__ bb, int* __restrict__ rois)
{
    const int t = threadIdx.x;
    if (t >= 96) return;
    const int scale = t >> 5;
    const int r = (t & 31) >> 2;
    const int j = t & 3;
    const float S[3]  = {122.0f, 63.0f, 33.0f};
    const float Hh[3] = {61.0f, 31.5f, 16.5f};
    const float s = S[scale], h = Hh[scale];
    const float b0 = bb[r * 4 + 0], b1 = bb[r * 4 + 1];
    const float b2 = bb[r * 4 + 2], b3 = bb[r * 4 + 3];
    float v;
    if      (j == 0) v = s * b0 - h * b2;
    else if (j == 1) v = s * b1 - h * b3;
    else if (j == 2) v = s * b0 + h * b2;
    else             v = s * b1 + h * b3;
    v = fminf(fmaxf(v, 0.0f), s);
    rois[t] = (int)v;
}

// ---------------------------------------------------------------------------
// Adaptive ROI maxpool -> (B, C, 8, 16), written into concatenated roi buffer
// ---------------------------------------------------------------------------
__global__ __launch_bounds__(256) void roipool_kernel(
    const float* __restrict__ feat, const int* __restrict__ roi,
    float* __restrict__ out, int B, int C, int Hf, int Wf, int Ctot, int coff)
{
    const int idx = blockIdx.x * blockDim.x + threadIdx.x;
    const int total = B * C * 128;
    if (idx >= total) return;
    const int ow = idx & 15;
    const int oh = (idx >> 4) & 7;
    const int c  = (idx >> 7) % C;
    const int b  = idx / (C * 128);

    const int x0 = roi[b * 4 + 0];
    const int y0 = roi[b * 4 + 1];
    const int x1 = max(min(roi[b * 4 + 2], Wf - 1), x0);
    const int y1 = max(min(roi[b * 4 + 3], Hf - 1), y0);
    const int H = y1 - y0 + 1;
    const int W = x1 - x0 + 1;
    int hs = y0 + (oh * H) / 8;
    int he = y0 + ((oh + 1) * H + 7) / 8;
    int xs = x0 + (ow * W) / 16;
    int xe = x0 + ((ow + 1) * W + 15) / 16;
    he = min(he, Hf);
    xe = min(xe, Wf);

    const float* __restrict__ p = feat + ((size_t)b * C + c) * Hf * Wf;
    float m = -__builtin_inff();
    for (int iy = hs; iy < he; ++iy)
        for (int ix = xs; ix < xe; ++ix)
            m = fmaxf(m, p[iy * Wf + ix]);
    out[((size_t)b * Ctot + (coff + c)) * 128 + oh * 16 + ow] = m;
}

// ---------------------------------------------------------------------------
// Host orchestration
// ---------------------------------------------------------------------------
extern "C" void kernel_launch(void* const* d_in, const int* in_sizes, int n_in,
                              void* d_out, int out_size, void* d_ws, size_t ws_size,
                              hipStream_t stream) {
    (void)in_sizes; (void)n_in; (void)out_size; (void)ws_size;

    const float* image = (const float*)d_in[0];
    const float* conv_w[10]; const float* conv_b[10];
    const float* bn_g[10];   const float* bn_b[10];
    const float* bn_mu[10];  const float* bn_var[10];
    for (int i = 0; i < 10; ++i) {
        conv_w[i] = (const float*)d_in[1 + i];
        conv_b[i] = (const float*)d_in[11 + i];
        bn_g[i]   = (const float*)d_in[21 + i];
        bn_b[i]   = (const float*)d_in[31 + i];
        bn_mu[i]  = (const float*)d_in[41 + i];
        bn_var[i] = (const float*)d_in[51 + i];
    }
    const float* cls_w[3] = {(const float*)d_in[61], (const float*)d_in[62], (const float*)d_in[63]};
    const float* cls_b[3] = {(const float*)d_in[64], (const float*)d_in[65], (const float*)d_in[66]};
    const float* head_w[7]; const float* head_b[7];
    for (int i = 0; i < 7; ++i) {
        head_w[i] = (const float*)d_in[67 + i];
        head_b[i] = (const float*)d_in[74 + i];
    }

    // ---- workspace carve-up (floats) ----
    float* ws = (float*)d_ws;
    size_t o = 0;
    float* buf0    = ws + o; o += 22118400;   // max conv out: 8*48*240*240
    float* buf1    = ws + o; o += 5622144;    // max ping pool out: 8*48*121*121
    float* feats1  = ws + o; o += 7620608;    // 8*64*122*122
    float* feats3  = ws + o; o += 5080320;    // 8*160*63*63
    float* feats5  = ws + o; o += 1672704;    // 8*192*33*33
    float* h0      = ws + o; o += 800;
    float* h1      = ws + o; o += 800;
    float* roiflat = ws + o; o += 425984;     // 8 * 416*128
    int*   rois    = (int*)(ws + o);          // 96 ints

    static const int CHs[11]  = {3, 48, 64, 128, 160, 192, 192, 192, 192, 192, 192};
    static const int KSs[10]  = {5, 5, 5, 5, 5, 5, 5, 5, 3, 3};
    static const int CSTs[10] = {2, 1, 1, 1, 1, 1, 1, 1, 1, 1};
    static const int PSTs[10] = {2, 1, 2, 1, 2, 1, 2, 1, 2, 1};

    const int B = 8;
    int Hin = 480, Win = 480;
    const float* cur = image;

    for (int i = 0; i < 10; ++i) {
        const int cin = CHs[i], cout = CHs[i + 1];
        const int k = KSs[i], cs = CSTs[i], ps = PSTs[i], pad = (k - 1) / 2;
        const int Hc = (Hin + 2 * pad - k) / cs + 1;
        const int Wc = (Win + 2 * pad - k) / cs + 1;
        const int Npix = B * Hc * Wc;
        const int ntiles = (Npix + 15) / 16;
        dim3 grid((ntiles + 7) / 8, cout / 16);
        if (k == 5 && cs == 2)
            conv_bn_relu_wmma<5, 2><<<grid, 256, 0, stream>>>(
                cur, conv_w[i], conv_b[i], bn_g[i], bn_b[i], bn_mu[i], bn_var[i],
                buf0, B, cin, Hin, Win, cout, Hc, Wc);
        else if (k == 5)
            conv_bn_relu_wmma<5, 1><<<grid, 256, 0, stream>>>(
                cur, conv_w[i], conv_b[i], bn_g[i], bn_b[i], bn_mu[i], bn_var[i],
                buf0, B, cin, Hin, Win, cout, Hc, Wc);
        else
            conv_bn_relu_wmma<3, 1><<<grid, 256, 0, stream>>>(
                cur, conv_w[i], conv_b[i], bn_g[i], bn_b[i], bn_mu[i], bn_var[i],
                buf0, B, cin, Hin, Win, cout, Hc, Wc);

        const int Hp = Hc / ps + 1;
        const int Wp = Wc / ps + 1;
        float* pdst = (i == 1) ? feats1 : (i == 3) ? feats3 : (i == 5) ? feats5 : buf1;
        const int tot = B * cout * Hp * Wp;
        maxpool2x2<<<(tot + 255) / 256, 256, 0, stream>>>(
            buf0, pdst, B * cout, Hc, Wc, Hp, Wp, ps);

        cur = pdst;
        Hin = Hp; Win = Wp;
    }

    // ---- classifier MLP: (8,23232) -> 100 -> 100 -> 4 (bb goes to d_out[0:32]) ----
    float* outf = (float*)d_out;
    fc_warp<<<(8 * 100 + 7) / 8, 256, 0, stream>>>(cur,  cls_w[0], cls_b[0], h0, 8, 23232, 100, 1);
    fc_warp<<<(8 * 100 + 7) / 8, 256, 0, stream>>>(h0,   cls_w[1], cls_b[1], h1, 8, 100,   100, 1);
    fc_warp<<<(8 * 4 + 7) / 8,   256, 0, stream>>>(h1,   cls_w[2], cls_b[2], outf, 8, 100, 4, 0);

    // ---- ROIs + adaptive maxpool into concatenated (8, 416, 8, 16) ----
    rois_kernel<<<1, 128, 0, stream>>>(outf, rois);
    roipool_kernel<<<(8 * 64  * 128 + 255) / 256, 256, 0, stream>>>(feats1, rois +  0, roiflat, 8, 64,  122, 122, 416, 0);
    roipool_kernel<<<(8 * 160 * 128 + 255) / 256, 256, 0, stream>>>(feats3, rois + 32, roiflat, 8, 160, 63,  63,  416, 64);
    roipool_kernel<<<(8 * 192 * 128 + 255) / 256, 256, 0, stream>>>(feats5, rois + 64, roiflat, 8, 192, 33,  33,  416, 224);

    // ---- 7 head GEMMs: (8, 53248) @ (53248, d) -> d_out ----
    static const int HD[7] = {38, 25, 35, 35, 35, 35, 35};
    int off = 32;
    for (int i = 0; i < 7; ++i) {
        fc_warp<<<(8 * HD[i] + 7) / 8, 256, 0, stream>>>(
            roiflat, head_w[i], head_b[i], outf + off, 8, 53248, HD[i], 0);
        off += 8 * HD[i];
    }
}
